// LSTMDegradationModel_79267916415505
// MI455X (gfx1250) — matrix operand order repaired and checked by reading, not compiled, not run on hardware
//
#include <hip/hip_runtime.h>
#include <hip/hip_bf16.h>

// ---------------------------------------------------------------------------
// Persistent fused 2-layer LSTM + head for MI455X (gfx1250, wave32, WMMA).
//  - f16 WMMA (v_wmma_f32_16x16x32_f16), f32 accumulation
//  - all weights converted to f16 ONCE into LDS, resident for all 1024 steps
//  - 64 persistent WGs (16 unit-strips x 4 row-groups), device-side time loop
//  - c-state resident in LDS; h-state exchanged via L2 with sw grid barrier
// ---------------------------------------------------------------------------

typedef __attribute__((ext_vector_type(16))) _Float16 v16h;
typedef __attribute__((ext_vector_type(8)))  _Float16 v8h;
typedef __attribute__((ext_vector_type(8)))  float    v8f;

#define Bsz   512
#define Tsz   1024
#define Dsz   64
#define Hsz   256
#define NBLK  64       // 16 unit-strips x 4 row-groups (persistent grid)
#define NTHR  256      // 8 wave32 per WG
#define ROWS  128      // batch rows per WG
#define UNITS 16       // hidden units per WG
#define NC    64       // gate columns per WG = 4*UNITS (i,f,g,o)
#define K0    320      // D + H   (layer 0 fused [x|h] K)
#define K1    512      // H + H   (layer 1 fused [y0|h1] K)
#define AS    520      // A LDS stride (halves), padded (mult of 8)
#define WS0   328      // W0 LDS stride (halves), padded
#define WS1   520      // W1 LDS stride (halves), padded
#define GS    68       // gates LDS stride (floats), padded

#define SMEM_HALVES (ROWS*AS + NC*WS0 + NC*WS1)
#define SMEM_FLOATS (ROWS*GS + 2*ROWS*UNITS + 2*NC + UNITS + ROWS)
#define SMEM_BYTES  (SMEM_HALVES*2 + SMEM_FLOATS*4 + ROWS*4)

__device__ __forceinline__ v16h cat16(v8h lo, v8h hi) {
  return __builtin_shufflevector(lo, hi, 0,1,2,3,4,5,6,7,8,9,10,11,12,13,14,15);
}
__device__ __forceinline__ float sigf(float x) { return 1.0f / (1.0f + __expf(-x)); }

// Sense-reversing software grid barrier (agent scope, L2-backed atomics).
__device__ __forceinline__ void grid_barrier(unsigned* cnt, unsigned* gen, int tid) {
  __threadfence();
  __syncthreads();
  if (tid == 0) {
    unsigned g = __hip_atomic_load(gen, __ATOMIC_ACQUIRE, __HIP_MEMORY_SCOPE_AGENT);
    unsigned a = __hip_atomic_fetch_add(cnt, 1u, __ATOMIC_ACQ_REL, __HIP_MEMORY_SCOPE_AGENT);
    if (a == (unsigned)(NBLK - 1)) {
      __hip_atomic_store(cnt, 0u, __ATOMIC_RELAXED, __HIP_MEMORY_SCOPE_AGENT);
      __hip_atomic_fetch_add(gen, 1u, __ATOMIC_RELEASE, __HIP_MEMORY_SCOPE_AGENT);
    } else {
      while (__hip_atomic_load(gen, __ATOMIC_ACQUIRE, __HIP_MEMORY_SCOPE_AGENT) == g) {
        __builtin_amdgcn_s_sleep(2);
      }
    }
  }
  __syncthreads();
  __threadfence();
}

// One (128 x 64) gate tile: C += A(128xK) * W^T slice, bias-initialized.
// Fragment layouts per CDNA5 ISA 7.12.2 (A 16x32 f16; B 32x16 f16; C 16x16 f32).
template <int KCH>
__device__ __forceinline__ void gemm_tile(const _Float16* A, const _Float16* W, int ws,
                                          const float* bias, float* G,
                                          int lane, int waveM, int waveN) {
  const int l  = lane & 15;
  const int hi = lane >> 4;
  v8f a00, a01, a10, a11;
  {
    float b0v = bias[waveN + l];
    float b1v = bias[waveN + 16 + l];
#pragma unroll
    for (int r = 0; r < 8; ++r) { a00[r] = b0v; a01[r] = b1v; a10[r] = b0v; a11[r] = b1v; }
  }
  const _Float16* ar0 = A + (waveM + l) * AS;
  const _Float16* ar1 = ar0 + 16 * AS;
  const _Float16* bc0 = W + (waveN + l) * ws + hi * 8;
  const _Float16* bc1 = bc0 + 16 * ws;
#pragma unroll
  for (int kc = 0; kc < KCH; ++kc) {
    const int kb = kc * 32;
    v16h fa0 = cat16(*(const v8h*)(ar0 + kb + hi * 8), *(const v8h*)(ar0 + kb + 16 + hi * 8));
    v16h fa1 = cat16(*(const v8h*)(ar1 + kb + hi * 8), *(const v8h*)(ar1 + kb + 16 + hi * 8));
    v16h fb0 = cat16(*(const v8h*)(bc0 + kb), *(const v8h*)(bc0 + kb + 16));
    v16h fb1 = cat16(*(const v8h*)(bc1 + kb), *(const v8h*)(bc1 + kb + 16));
    a00 = __builtin_amdgcn_wmma_f32_16x16x32_f16(false, fa0, false, fb0, (short)0, a00, false, false);
    a01 = __builtin_amdgcn_wmma_f32_16x16x32_f16(false, fa0, false, fb1, (short)0, a01, false, false);
    a10 = __builtin_amdgcn_wmma_f32_16x16x32_f16(false, fa1, false, fb0, (short)0, a10, false, false);
    a11 = __builtin_amdgcn_wmma_f32_16x16x32_f16(false, fa1, false, fb1, (short)0, a11, false, false);
  }
  const int Mb = waveM + (hi << 3);
  const int Nb = waveN + l;
#pragma unroll
  for (int r = 0; r < 8; ++r) {
    G[(Mb + r)      * GS + Nb]      = a00[r];
    G[(Mb + r)      * GS + Nb + 16] = a01[r];
    G[(Mb + 16 + r) * GS + Nb]      = a10[r];
    G[(Mb + 16 + r) * GS + Nb + 16] = a11[r];
  }
}

__global__ void zero_bar_kernel(unsigned* p) {
  if (threadIdx.x < 2) p[threadIdx.x] = 0u;
}

__global__ __launch_bounds__(NTHR) void lstm_persistent_kernel(
    const float* __restrict__ xg,   const int* __restrict__ leng,
    const float* __restrict__ wih0, const float* __restrict__ whh0,
    const float* __restrict__ bih0, const float* __restrict__ bhh0,
    const float* __restrict__ wih1, const float* __restrict__ whh1,
    const float* __restrict__ bih1, const float* __restrict__ bhh1,
    const float* __restrict__ headw, const float* __restrict__ headb,
    float* __restrict__ outp,
    _Float16* __restrict__ h0g, _Float16* __restrict__ h1g,
    _Float16* __restrict__ y0g, float* __restrict__ hpart,
    unsigned* __restrict__ barcnt, unsigned* __restrict__ bargen) {
  extern __shared__ char smem_raw[];
  _Float16* Alds  = (_Float16*)smem_raw;             // [ROWS][AS]
  _Float16* W0lds = Alds + ROWS * AS;                // [NC][WS0]
  _Float16* W1lds = W0lds + NC * WS0;                // [NC][WS1]
  float*    Glds  = (float*)(W1lds + NC * WS1);      // [ROWS][GS]
  float*    C0    = Glds + ROWS * GS;                // [ROWS*UNITS] cell state L0
  float*    C1    = C0 + ROWS * UNITS;               // cell state L1
  float*    B0b   = C1 + ROWS * UNITS;               // [NC] combined bias L0
  float*    B1b   = B0b + NC;                        // [NC] combined bias L1
  float*    HW    = B1b + NC;                        // [UNITS] head weights
  float*    HP    = HW + UNITS;                      // [ROWS] head partials
  int*      LEN   = (int*)(HP + ROWS);               // [ROWS]

  const int tid    = threadIdx.x;
  const int wg     = blockIdx.x;
  const int jstrip = wg & 15;            // which 16-unit strip
  const int j0     = jstrip * UNITS;
  const int m0     = (wg >> 4) * ROWS;   // which 128-row group
  const int lane   = tid & 31;
  const int wave   = tid >> 5;
  const int waveM  = (wave & 3) * 32;
  const int waveN  = (wave >> 2) * 32;
  const float headbias = headb[0];

  // ---- one-time setup: weights -> f16 LDS (resident for all timesteps) ----
  for (int idx = tid; idx < NC * K0; idx += NTHR) {
    int n = idx / K0, k = idx - n * K0;
    int r = ((n >> 4) << 8) + j0 + (n & 15);         // gate*256 + unit
    float w = (k < Dsz) ? wih0[r * Dsz + k] : whh0[r * Hsz + (k - Dsz)];
    W0lds[n * WS0 + k] = (_Float16)w;
  }
  for (int idx = tid; idx < NC * K1; idx += NTHR) {
    int n = idx >> 9, k = idx & (K1 - 1);
    int r = ((n >> 4) << 8) + j0 + (n & 15);
    float w = (k < Hsz) ? wih1[r * Hsz + k] : whh1[r * Hsz + (k - Hsz)];
    W1lds[n * WS1 + k] = (_Float16)w;
  }
  if (tid < NC) {
    int r = ((tid >> 4) << 8) + j0 + (tid & 15);
    B0b[tid] = bih0[r] + bhh0[r];
    B1b[tid] = bih1[r] + bhh1[r];
  }
  if (tid < UNITS) HW[tid] = headw[j0 + tid];
  if (tid < ROWS)  LEN[tid] = leng[m0 + tid];
  for (int idx = tid; idx < ROWS * UNITS; idx += NTHR) {
    C0[idx] = 0.0f;
    C1[idx] = 0.0f;
    int row = idx & (ROWS - 1), u = idx >> 7;
    size_t g = (size_t)(m0 + row) * Hsz + (j0 + u);
    h0g[g] = (_Float16)0.0f;
    h1g[g] = (_Float16)0.0f;
  }
  grid_barrier(barcnt, bargen, tid);

  // ---- device-side recurrence over T ----
  for (int t = 0; t < Tsz; ++t) {
    // stage A = [x_t (f32->f16) | h0] into LDS
    for (int idx = tid; idx < ROWS * Dsz; idx += NTHR) {
      int row = idx >> 6, d = idx & (Dsz - 1);
      Alds[row * AS + d] =
          (_Float16)xg[(size_t)(m0 + row) * (Tsz * Dsz) + (size_t)t * Dsz + d];
    }
    for (int idx = tid; idx < ROWS * (Hsz / 8); idx += NTHR) {
      int row = idx >> 5, jc = idx & 31;
      *(v8h*)(Alds + row * AS + Dsz + jc * 8) =
          *(const v8h*)(h0g + (size_t)(m0 + row) * Hsz + jc * 8);
    }
    // deterministic head reduction for step t-1 (published last step)
    if (t > 0 && jstrip == 0 && tid < ROWS) {
      int b = m0 + tid;
      float s = headbias;
#pragma unroll
      for (int st = 0; st < 16; ++st) s += hpart[st * Bsz + b];
      outp[(size_t)b * Tsz + (t - 1)] = s;
    }
    __syncthreads();
    gemm_tile<K0 / 32>(Alds, W0lds, WS0, B0b, Glds, lane, waveM, waveN);
    __syncthreads();
    // layer-0 pointwise: i,f,g,o -> c0,h0 ; packed-seq mask; y0 zero-padded
    for (int idx = tid; idx < ROWS * UNITS; idx += NTHR) {
      int row = idx & (ROWS - 1), u = idx >> 7;
      bool msk = t < LEN[row];
      const float* gr = Glds + row * GS;
      float iv = sigf(gr[u]);
      float fv = sigf(gr[16 + u]);
      float gv = tanhf(gr[32 + u]);
      float ov = sigf(gr[48 + u]);
      float cn = fv * C0[idx] + iv * gv;
      float hn = ov * tanhf(cn);
      if (msk) C0[idx] = cn;
      size_t g = (size_t)(m0 + row) * Hsz + (j0 + u);
      _Float16 hh = (_Float16)hn;
      if (msk) h0g[g] = hh;
      y0g[g] = msk ? hh : (_Float16)0.0f;   // always store (pad -> 0)
    }
    grid_barrier(barcnt, bargen, tid);

    // stage A = [y0_t | h1]
    for (int idx = tid; idx < ROWS * (Hsz / 8); idx += NTHR) {
      int row = idx >> 5, jc = idx & 31;
      *(v8h*)(Alds + row * AS + jc * 8) =
          *(const v8h*)(y0g + (size_t)(m0 + row) * Hsz + jc * 8);
      *(v8h*)(Alds + row * AS + Hsz + jc * 8) =
          *(const v8h*)(h1g + (size_t)(m0 + row) * Hsz + jc * 8);
    }
    if (tid < ROWS) HP[tid] = 0.0f;
    __syncthreads();
    gemm_tile<K1 / 32>(Alds, W1lds, WS1, B1b, Glds, lane, waveM, waveN);
    __syncthreads();
    // layer-1 pointwise + local head partial (16 units of the 256-dot)
    for (int idx = tid; idx < ROWS * UNITS; idx += NTHR) {
      int row = idx & (ROWS - 1), u = idx >> 7;
      bool msk = t < LEN[row];
      const float* gr = Glds + row * GS;
      float iv = sigf(gr[u]);
      float fv = sigf(gr[16 + u]);
      float gv = tanhf(gr[32 + u]);
      float ov = sigf(gr[48 + u]);
      float cn = fv * C1[idx] + iv * gv;
      float hn = ov * tanhf(cn);
      if (msk) {
        C1[idx] = cn;
        h1g[(size_t)(m0 + row) * Hsz + (j0 + u)] = (_Float16)hn;
        atomicAdd(&HP[row], hn * HW[u]);   // ds_add_f32, y1 masked -> 0 contribution
      }
    }
    __syncthreads();
    if (tid < ROWS) hpart[jstrip * Bsz + (m0 + tid)] = HP[tid];
    grid_barrier(barcnt, bargen, tid);
  }

  // final head output for t = T-1
  if (jstrip == 0 && tid < ROWS) {
    int b = m0 + tid;
    float s = headbias;
#pragma unroll
    for (int st = 0; st < 16; ++st) s += hpart[st * Bsz + b];
    outp[(size_t)b * Tsz + (Tsz - 1)] = s;
  }
}

extern "C" void kernel_launch(void* const* d_in, const int* in_sizes, int n_in,
                              void* d_out, int out_size, void* d_ws, size_t ws_size,
                              hipStream_t stream) {
  (void)in_sizes; (void)n_in; (void)out_size; (void)ws_size;
  const float* xg    = (const float*)d_in[0];
  const int*   leng  = (const int*)d_in[1];
  const float* wih0  = (const float*)d_in[2];
  const float* whh0  = (const float*)d_in[3];
  const float* bih0  = (const float*)d_in[4];
  const float* bhh0  = (const float*)d_in[5];
  const float* wih1  = (const float*)d_in[6];
  const float* whh1  = (const float*)d_in[7];
  const float* bih1  = (const float*)d_in[8];
  const float* bhh1  = (const float*)d_in[9];
  const float* headw = (const float*)d_in[10];
  const float* headb = (const float*)d_in[11];
  float* outp = (float*)d_out;

  // workspace carve: [barrier words | h0 | h1 | y0 (f16, B*H each) | head partials]
  unsigned* barwords = (unsigned*)d_ws;
  _Float16* h0g = (_Float16*)((char*)d_ws + 256);
  _Float16* h1g = h0g + (size_t)Bsz * Hsz;
  _Float16* y0g = h1g + (size_t)Bsz * Hsz;
  float* hpart  = (float*)(y0g + (size_t)Bsz * Hsz);   // [16][Bsz]

  (void)hipFuncSetAttribute((const void*)lstm_persistent_kernel,
                            hipFuncAttributeMaxDynamicSharedMemorySize, SMEM_BYTES);

  zero_bar_kernel<<<1, 32, 0, stream>>>(barwords);
  lstm_persistent_kernel<<<dim3(NBLK), dim3(NTHR), SMEM_BYTES, stream>>>(
      xg, leng, wih0, whh0, bih0, bhh0, wih1, whh1, bih1, bhh1, headw, headb,
      outp, h0g, h1g, y0g, hpart, barwords, barwords + 1);
}
